// LinearAttentionBlock_3874060501406
// MI455X (gfx1250) — matrix operand order repaired and checked by reading, not compiled, not run on hardware
//
#include <hip/hip_runtime.h>
#include <math.h>

typedef __attribute__((ext_vector_type(16))) _Float16 v16h;
typedef __attribute__((ext_vector_type(8)))  _Float16 v8h;
typedef __attribute__((ext_vector_type(8)))  float    v8f;
typedef __attribute__((ext_vector_type(4)))  float    v4f;

#define D_MODEL 128
#define FF_H    512
#define NTOK    (16L*8192L)
#define LN_EPS  1e-5f

static __device__ inline v8f wmma_f16(v16h a, v16h b, v8f c) {
  // D = A(16x32 f16) * B(32x16 f16) + C(16x16 f32)
  return __builtin_amdgcn_wmma_f32_16x16x32_f16(false, a, false, b, (short)0, c, false, false);
}

// ---- operand loaders (ISA 7.12.2 layouts, wave32) ----
// A 16x32 f16 from row-major f32: lane<16 holds M=lane, K = k0+[0..7], k0+16+[0..7]; lane>=16 K += 8.
static __device__ inline v16h load_A_g32(const float* __restrict__ X, int ld, long m0, int k0, int lane) {
  long m = m0 + (lane & 15);
  int  g = (lane >> 4) & 1;
  const float* p = X + m * (long)ld + k0 + 8 * g;
  v4f a0 = *(const v4f*)(p);
  v4f a1 = *(const v4f*)(p + 4);
  v4f a2 = *(const v4f*)(p + 16);
  v4f a3 = *(const v4f*)(p + 20);
  v16h a;
#pragma unroll
  for (int i = 0; i < 4; i++) {
    a[i]      = (_Float16)a0[i];
    a[i + 4]  = (_Float16)a1[i];
    a[i + 8]  = (_Float16)a2[i];
    a[i + 12] = (_Float16)a3[i];
  }
  return a;
}

static __device__ inline v16h load_A_g16(const _Float16* __restrict__ X, int ld, long m0, int k0, int lane) {
  long m = m0 + (lane & 15);
  int  g = (lane >> 4) & 1;
  const _Float16* p = X + m * (long)ld + k0 + 8 * g;
  v8h lo = *(const v8h*)p;
  v8h hi = *(const v8h*)(p + 16);
  v16h a;
#pragma unroll
  for (int i = 0; i < 8; i++) { a[i] = lo[i]; a[i + 8] = hi[i]; }
  return a;
}

static __device__ inline v16h load_A_lds32(const float* Xs, int ld, int k0, int lane) {
  int m = lane & 15;
  int g = (lane >> 4) & 1;
  const float* p = Xs + m * ld + k0 + 8 * g;
  v4f a0 = *(const v4f*)(p);
  v4f a1 = *(const v4f*)(p + 4);
  v4f a2 = *(const v4f*)(p + 16);
  v4f a3 = *(const v4f*)(p + 20);
  v16h a;
#pragma unroll
  for (int i = 0; i < 4; i++) {
    a[i]      = (_Float16)a0[i];
    a[i + 4]  = (_Float16)a1[i];
    a[i + 8]  = (_Float16)a2[i];
    a[i + 12] = (_Float16)a3[i];
  }
  return a;
}

static __device__ inline v16h load_A_lds16(const _Float16* Xs, int ld, int k0, int lane) {
  int m = lane & 15;
  int g = (lane >> 4) & 1;
  const _Float16* p = Xs + m * ld + k0 + 8 * g;
  v8h lo = *(const v8h*)p;
  v8h hi = *(const v8h*)(p + 16);
  v16h a;
#pragma unroll
  for (int i = 0; i < 8; i++) { a[i] = lo[i]; a[i + 8] = hi[i]; }
  return a;
}

// B 32x16 f16 from PRE-TRANSPOSED f16 weight Wt[N][K] (ldk = K):
// lane<16: N = n0+lane, K = k0+[0..7], k0+16+[0..7]; lane>=16: K += 8. -> 2x global_load_b128.
static __device__ inline v16h load_B_t16(const _Float16* __restrict__ Wt, int ldk, int k0, int n0, int lane) {
  int col = n0 + (lane & 15);
  int kb  = k0 + 8 * ((lane >> 4) & 1);
  const _Float16* p = Wt + (long)col * ldk + kb;
  v8h lo = *(const v8h*)p;
  v8h hi = *(const v8h*)(p + 16);
  v16h b;
#pragma unroll
  for (int i = 0; i < 8; i++) { b[i] = lo[i]; b[i + 8] = hi[i]; }
  return b;
}

// ============ prep: transpose f32 [K][N] -> f16 [N][K] ============
__global__ __launch_bounds__(256) void transpose_f16_kernel(
    const float* __restrict__ src, _Float16* __restrict__ dst, int K, int N) {
  int idx = blockIdx.x * 256 + threadIdx.x;   // idx = n*K + k
  if (idx >= K * N) return;
  int n = idx / K;
  int k = idx - n * K;
  dst[idx] = (_Float16)src[(long)k * N + n];
}

// ============ kernel 1: QKV projection + phi, f16 output ============
template <bool PHI>
static __device__ inline void proj16(const v16h A[4], const _Float16* __restrict__ Wt,
                                     _Float16* __restrict__ O, long tok0, int lane) {
  int col = lane & 15, grp = (lane >> 4) & 1;
#pragma unroll
  for (int nt = 0; nt < 8; nt++) {
    v8f c = {};
#pragma unroll
    for (int ks = 0; ks < 4; ks++)
      c = wmma_f16(A[ks], load_B_t16(Wt, D_MODEL, ks * 32, nt * 16, lane), c);
    int n = nt * 16 + col;
#pragma unroll
    for (int r = 0; r < 8; r++) {
      float v = c[r];
      if (PHI) v = (v > 0.f) ? (v + 1.f) : __expf(v);  // phi = elu + 1
      O[(tok0 + r + 8 * grp) * D_MODEL + n] = (_Float16)v;
    }
  }
}

__global__ __launch_bounds__(256) void qkv_phi_kernel(
    const float* __restrict__ x,
    const _Float16* __restrict__ WqT, const _Float16* __restrict__ WkT,
    const _Float16* __restrict__ WvT,
    _Float16* __restrict__ Qb, _Float16* __restrict__ Kb, _Float16* __restrict__ Vb) {
  int lane = threadIdx.x & 31;
  int wave = threadIdx.x >> 5;
  long tok0 = ((long)blockIdx.x * 8 + wave) * 16;

  v16h A[4];
#pragma unroll
  for (int ks = 0; ks < 4; ks++) A[ks] = load_A_g32(x, D_MODEL, tok0, ks * 32, lane);

  proj16<true>(A, WqT, Qb, tok0, lane);
  proj16<true>(A, WkT, Kb, tok0, lane);
  proj16<false>(A, WvT, Vb, tok0, lane);
}

// ============ kernel 2: KV = K^T V (32x32 per B,H) and Ksum ============
__global__ __launch_bounds__(256) void kv_reduce_kernel(
    const _Float16* __restrict__ Kb, const _Float16* __restrict__ Vb,
    float* __restrict__ KV, float* __restrict__ Ksum) {
  int lane = threadIdx.x & 31;      // lane = m (V column) and d (for Ksum)
  int wave = threadIdx.x >> 5;      // token segment
  int pair = blockIdx.x;            // b*4 + h, 64 pairs
  int b = pair >> 2, h = pair & 3;
  long base = (long)b * 8192 * D_MODEL + h * 32;
  long t0 = (long)wave * 1024;

  float kv[32];
#pragma unroll
  for (int d = 0; d < 32; d++) kv[d] = 0.f;
  float ks = 0.f;

  for (long t = t0; t < t0 + 1024; t++) {
    const _Float16* krow = Kb + base + t * D_MODEL;
    const _Float16* vrow = Vb + base + t * D_MODEL;
    float kd = (float)krow[lane];
    float vm = (float)vrow[lane];
    ks += kd;
#pragma unroll
    for (int d = 0; d < 32; d++) {
      float kdd = __shfl(kd, d, 32);
      kv[d] = fmaf(kdd, vm, kv[d]);
    }
  }
  float* KVp = KV + (long)pair * 1024;  // [d][m] row-major, ld=32
#pragma unroll
  for (int d = 0; d < 32; d++) atomicAdd(&KVp[d * 32 + lane], kv[d]);
  atomicAdd(&Ksum[pair * 32 + lane], ks);
}

// ============ kernel 2b: KV f32 [pair][d][m] -> f16 transposed+scaled [pair][m][d] ============
__global__ __launch_bounds__(256) void kv_finalize_kernel(
    const float* __restrict__ KV, _Float16* __restrict__ KVt) {
  int idx = blockIdx.x * 256 + threadIdx.x;   // 64*32*32 = 65536
  int pair = idx >> 10;
  int m = (idx >> 5) & 31;
  int d = idx & 31;
  KVt[idx] = (_Float16)(KV[(pair << 10) + (d << 5) + m] * (1.0f / 8192.0f));
}

// ============ kernel 3: fused attn·KV/denom, @Wo, +res, LN1, FFN, +res, LN2 ============
__global__ __launch_bounds__(64) void attn_ffn_kernel(
    const float* __restrict__ x, const _Float16* __restrict__ Qb,
    const _Float16* __restrict__ KVt, const float* __restrict__ Ksum,
    const _Float16* __restrict__ WoT,
    const float* __restrict__ ln1g, const float* __restrict__ ln1b,
    const _Float16* __restrict__ W1T, const float* __restrict__ b1,
    const _Float16* __restrict__ W2T, const float* __restrict__ b2,
    const float* __restrict__ ln2g, const float* __restrict__ ln2b,
    float* __restrict__ out) {
  __shared__ __align__(16) float    sA[2][16 * D_MODEL];   // attn, later x2 (post-LN1)
  __shared__ __align__(16) float    sY[2][16 * D_MODEL];   // pre-LN sums
  __shared__ __align__(16) _Float16 sH[2][16 * FF_H];      // relu hidden (f16)

  int lane = threadIdx.x & 31;
  int wave = threadIdx.x >> 5;
  float*    bufA = sA[wave];
  float*    bufY = sY[wave];
  _Float16* bufH = sH[wave];

  long tok0 = ((long)blockIdx.x * 2 + wave) * 16;
  int  bidx = (int)(tok0 >> 13);             // batch index (8192 tokens per batch)
  int  col = lane & 15, grp = (lane >> 4) & 1;
  const float scale = 1.0f / 8192.0f;        // KVt is pre-scaled; cancels in the division

  // ---- numerator: Q (16x32 per head) @ scaled KV^T (B operand, k=d contiguous) ----
#pragma unroll
  for (int h = 0; h < 4; h++) {
    v16h aq = load_A_g16(Qb, D_MODEL, tok0, h * 32, lane);
    const _Float16* kvh = KVt + (long)(bidx * 4 + h) * 1024;   // [m][d], ldk=32
#pragma unroll
    for (int nt = 0; nt < 2; nt++) {
      v8f c = {};
      c = wmma_f16(aq, load_B_t16(kvh, 32, 0, nt * 16, lane), c);
      int cc = h * 32 + nt * 16 + col;
#pragma unroll
      for (int r = 0; r < 8; r++) bufA[(r + 8 * grp) * D_MODEL + cc] = c[r];
    }
  }
  __syncthreads();

  // ---- denominator + divide (lane owns row=lane&15; grp selects head pair) ----
  {
    int row = lane & 15;
    const _Float16* qrow = Qb + (tok0 + row) * D_MODEL;
#pragma unroll
    for (int hi = 0; hi < 2; hi++) {
      int hh = 2 * grp + hi;
      const float* kss = Ksum + (long)(bidx * 4 + hh) * 32;
      float dd = 0.f;
#pragma unroll
      for (int j = 0; j < 32; j++) dd = fmaf((float)qrow[hh * 32 + j], kss[j], dd);
      float inv = 1.0f / (fmaxf(dd, 1e-6f) * scale);
#pragma unroll
      for (int j = 0; j < 32; j++) bufA[row * D_MODEL + hh * 32 + j] *= inv;
    }
  }
  __syncthreads();

  // ---- attn @ Wo + residual ----
  {
    v16h Aop[4];
#pragma unroll
    for (int ks = 0; ks < 4; ks++) Aop[ks] = load_A_lds32(bufA, D_MODEL, ks * 32, lane);
#pragma unroll
    for (int nt = 0; nt < 8; nt++) {
      v8f c = {};
#pragma unroll
      for (int ks = 0; ks < 4; ks++)
        c = wmma_f16(Aop[ks], load_B_t16(WoT, D_MODEL, ks * 32, nt * 16, lane), c);
      int n = nt * 16 + col;
#pragma unroll
      for (int r = 0; r < 8; r++) {
        int m = r + 8 * grp;
        bufY[m * D_MODEL + n] = c[r] + x[(tok0 + m) * D_MODEL + n];
      }
    }
  }
  __syncthreads();

  // ---- LN1 -> x2 into bufA (row = lane&15, lane pair splits the 128 cols) ----
  {
    int row = lane & 15;
    int c0  = 64 * grp;
    float s = 0.f;
#pragma unroll
    for (int j = 0; j < 64; j++) s += bufY[row * D_MODEL + c0 + j];
    s += __shfl_xor(s, 16, 32);
    float mu = s * (1.0f / D_MODEL);
    float v = 0.f;
#pragma unroll
    for (int j = 0; j < 64; j++) {
      float d = bufY[row * D_MODEL + c0 + j] - mu;
      v = fmaf(d, d, v);
    }
    v += __shfl_xor(v, 16, 32);
    float rs = rsqrtf(v * (1.0f / D_MODEL) + LN_EPS);
#pragma unroll
    for (int j = 0; j < 64; j++) {
      int jj = c0 + j;
      bufA[row * D_MODEL + jj] = (bufY[row * D_MODEL + jj] - mu) * rs * ln1g[jj] + ln1b[jj];
    }
  }
  __syncthreads();

  // ---- FF1: relu(x2 @ W1 + b1) -> bufH (f16) ----
  {
    v16h Aop[4];
#pragma unroll
    for (int ks = 0; ks < 4; ks++) Aop[ks] = load_A_lds32(bufA, D_MODEL, ks * 32, lane);
    for (int nt = 0; nt < 32; nt++) {
      v8f c = {};
#pragma unroll
      for (int ks = 0; ks < 4; ks++)
        c = wmma_f16(Aop[ks], load_B_t16(W1T, D_MODEL, ks * 32, nt * 16, lane), c);
      int n = nt * 16 + col;
      float bias = b1[n];
#pragma unroll
      for (int r = 0; r < 8; r++) {
        float v = fmaxf(c[r] + bias, 0.f);
        bufH[(r + 8 * grp) * FF_H + n] = (_Float16)v;
      }
    }
  }
  __syncthreads();

  // ---- FF2: h @ W2 + b2 + x2 -> bufY  (ks outer: load each A once, 8 live accumulators) ----
  {
    v8f zero = {};
    v8f acc[8];
#pragma unroll
    for (int nt = 0; nt < 8; nt++) acc[nt] = zero;
    for (int ks = 0; ks < 16; ks++) {
      v16h a = load_A_lds16(bufH, FF_H, ks * 32, lane);
#pragma unroll
      for (int nt = 0; nt < 8; nt++)
        acc[nt] = wmma_f16(a, load_B_t16(W2T, FF_H, ks * 32, nt * 16, lane), acc[nt]);
    }
#pragma unroll
    for (int nt = 0; nt < 8; nt++) {
      int n = nt * 16 + col;
      float bias = b2[n];
#pragma unroll
      for (int r = 0; r < 8; r++) {
        int m = r + 8 * grp;
        bufY[m * D_MODEL + n] = acc[nt][r] + bias + bufA[m * D_MODEL + n];
      }
    }
  }
  __syncthreads();

  // ---- LN2 -> out (same split-row scheme) ----
  {
    int row = lane & 15;
    int c0  = 64 * grp;
    float s = 0.f;
#pragma unroll
    for (int j = 0; j < 64; j++) s += bufY[row * D_MODEL + c0 + j];
    s += __shfl_xor(s, 16, 32);
    float mu = s * (1.0f / D_MODEL);
    float v = 0.f;
#pragma unroll
    for (int j = 0; j < 64; j++) {
      float d = bufY[row * D_MODEL + c0 + j] - mu;
      v = fmaf(d, d, v);
    }
    v += __shfl_xor(v, 16, 32);
    float rs = rsqrtf(v * (1.0f / D_MODEL) + LN_EPS);
    float* orow = out + (tok0 + row) * D_MODEL;
#pragma unroll
    for (int j = 0; j < 64; j++) {
      int jj = c0 + j;
      orow[jj] = (bufY[row * D_MODEL + jj] - mu) * rs * ln2g[jj] + ln2b[jj];
    }
  }
}

extern "C" void kernel_launch(void* const* d_in, const int* in_sizes, int n_in,
                              void* d_out, int out_size, void* d_ws, size_t ws_size,
                              hipStream_t stream) {
  const float* x    = (const float*)d_in[0];
  const float* Wq   = (const float*)d_in[1];
  const float* Wk   = (const float*)d_in[2];
  const float* Wv   = (const float*)d_in[3];
  const float* Wo   = (const float*)d_in[4];
  const float* ln1g = (const float*)d_in[5];
  const float* ln1b = (const float*)d_in[6];
  const float* W1   = (const float*)d_in[7];
  const float* b1   = (const float*)d_in[8];
  const float* W2   = (const float*)d_in[9];
  const float* b2   = (const float*)d_in[10];
  const float* ln2g = (const float*)d_in[11];
  const float* ln2b = (const float*)d_in[12];
  float* out = (float*)d_out;

  // workspace layout
  _Float16* Qb  = (_Float16*)d_ws;
  _Float16* Kb  = Qb + NTOK * D_MODEL;
  _Float16* Vb  = Kb + NTOK * D_MODEL;
  float*    KV  = (float*)(Vb + NTOK * D_MODEL);   // [64][32][32] f32 accumulators
  float*    Ks  = KV + 64 * 32 * 32;               // [64][32]
  _Float16* KVt = (_Float16*)(Ks + 64 * 32);       // [64][32][32] f16, transposed+scaled
  _Float16* WqT = KVt + 64 * 32 * 32;              // [128][128]
  _Float16* WkT = WqT + D_MODEL * D_MODEL;
  _Float16* WvT = WkT + D_MODEL * D_MODEL;
  _Float16* WoT = WvT + D_MODEL * D_MODEL;
  _Float16* W1T = WoT + D_MODEL * D_MODEL;         // [512][128]
  _Float16* W2T = W1T + D_MODEL * FF_H;            // [128][512]

  hipMemsetAsync(KV, 0, (64 * 32 * 32 + 64 * 32) * sizeof(float), stream);

  // prep: f16 transposed weights (Wt[n][k] = W[k][n])
  transpose_f16_kernel<<<(D_MODEL * D_MODEL + 255) / 256, 256, 0, stream>>>(Wq, WqT, D_MODEL, D_MODEL);
  transpose_f16_kernel<<<(D_MODEL * D_MODEL + 255) / 256, 256, 0, stream>>>(Wk, WkT, D_MODEL, D_MODEL);
  transpose_f16_kernel<<<(D_MODEL * D_MODEL + 255) / 256, 256, 0, stream>>>(Wv, WvT, D_MODEL, D_MODEL);
  transpose_f16_kernel<<<(D_MODEL * D_MODEL + 255) / 256, 256, 0, stream>>>(Wo, WoT, D_MODEL, D_MODEL);
  transpose_f16_kernel<<<(D_MODEL * FF_H + 255) / 256, 256, 0, stream>>>(W1, W1T, D_MODEL, FF_H);
  transpose_f16_kernel<<<(FF_H * D_MODEL + 255) / 256, 256, 0, stream>>>(W2, W2T, FF_H, D_MODEL);

  // 131072 tokens / 16 per wave / 8 waves per block = 1024 blocks
  qkv_phi_kernel<<<1024, 256, 0, stream>>>(x, WqT, WkT, WvT, Qb, Kb, Vb);

  // 64 (B,H) pairs, 8 segment-waves each
  kv_reduce_kernel<<<64, 256, 0, stream>>>(Kb, Vb, KV, Ks);
  kv_finalize_kernel<<<256, 256, 0, stream>>>(KV, KVt);

  // 8192 tiles / 2 waves per block = 4096 blocks
  attn_ffn_kernel<<<4096, 64, 0, stream>>>(x, Qb, KVt, Ks, WoT, ln1g, ln1b,
                                           W1T, b1, W2T, b2, ln2g, ln2b, out);
}
